// AdditiveCouplingLayer_72696616452701
// MI455X (gfx1250) — compile-verified, matrix operand order checked by reading, not run on hardware
//
#include <hip/hip_runtime.h>
#include <hip/hip_bf16.h>

// ---------------------------------------------------------------------------
// AdditiveCouplingLayer for MI455X (gfx1250, wave32, WMMA).
// Pipeline (all on `stream`, graph-capture safe):
//   pack_x -> pack_w(x7) -> gemm<0> -> convert(id)
//          -> 5x [ zero_stats -> gemm<1>(relu+stats) -> convert(BN) ]
//          -> gemm<2> (bias + residual into interleaved output)
// GEMMs: v_wmma_f32_16x16x32_bf16, 64x64 wave tile, 128x256 block tile,
// compile-time LDA/K/NB so all loads are base+immediate-offset.
// ---------------------------------------------------------------------------

typedef __attribute__((ext_vector_type(16))) __bf16 v16bf;
typedef __attribute__((ext_vector_type(8)))  __bf16 v8bf;
typedef __attribute__((ext_vector_type(4)))  __bf16 v4bf;
typedef __attribute__((ext_vector_type(8)))  float  v8f;

#define BATCH   16384
#define DIN     784
#define HALF_D  392
#define HID     1024
#define K0PAD   416      // 392 padded to multiple of 32
#define NOUTPAD 512      // 392 padded to multiple of 256 (block tile N)

// ---------------------------------------------------------------------------
// 1) Gather even columns: write them straight to the output (x1 passthrough)
//    and store zero-padded bf16 copy as the first GEMM's A matrix.
// ---------------------------------------------------------------------------
__global__ __launch_bounds__(256) void pack_x_kernel(
    const float* __restrict__ x, float* __restrict__ out, __bf16* __restrict__ A)
{
    int t = blockIdx.x * blockDim.x + threadIdx.x;   // 16384 * 416 threads
    int row = t / K0PAD;
    int c   = t - row * K0PAD;
    if (row >= BATCH) return;
    if (c < HALF_D) {
        float xe = x[(size_t)row * DIN + 2 * c];
        out[(size_t)row * DIN + 2 * c] = xe;
        A[(size_t)row * K0PAD + c] = (__bf16)xe;
    } else {
        A[(size_t)row * K0PAD + c] = (__bf16)0.0f;   // K padding
    }
}

// ---------------------------------------------------------------------------
// 2) Pack a KxN fp32 weight matrix into bf16 WMMA B-fragments.
//    Fragment (kb, nb) covers K rows [kb*32, kb*32+32) x N cols [nb*16, nb*16+16).
//    Per-lane layout mirrors the ISA 16-bit operand layout:
//      lanes 0-15 : N = nb*16+lane,  K = {k..k+7, k+16..k+23}
//      lanes 16-31: N = nb*16+ln-16, K = {k+8..k+15, k+24..k+31}
//    Each lane's 16 bf16 are stored contiguously (32 B) -> single vector load.
// ---------------------------------------------------------------------------
__global__ __launch_bounds__(256) void pack_w_kernel(
    const float* __restrict__ W, __bf16* __restrict__ dst,
    int K, int N, int KB, int NB)          // KB = Kpad/32, NB = Npad/16
{
    int t = blockIdx.x * blockDim.x + threadIdx.x;
    int frag = t >> 5;
    int lane = t & 31;
    if (frag >= KB * NB) return;
    int kb = frag / NB;
    int nb = frag - kb * NB;
    int n  = nb * 16 + (lane & 15);
    int kbase = kb * 32 + ((lane >> 4) ? 8 : 0);
    __bf16* o = dst + ((size_t)frag << 9) + lane * 16;
#pragma unroll
    for (int j = 0; j < 16; ++j) {
        int k = kbase + (j < 8 ? j : 8 + j);
        float v = (k < K && n < N) ? W[(size_t)k * N + n] : 0.0f;
        o[j] = (__bf16)v;
    }
}

__global__ void zero_stats_kernel(float* __restrict__ stats)
{
    int t = blockIdx.x * blockDim.x + threadIdx.x;
    if (t < 2 * HID) stats[t] = 0.0f;
}

// ---------------------------------------------------------------------------
// 3) WMMA GEMM. Block = 256 threads = 8 wave32s (2 waveM x 4 waveN),
//    block tile 128x256, wave tile 64x64 (4x4 16x16 accumulators).
//    Per k-step: 16 b128 loads, 16 WMMAs, 2 pointer increments.
//    MODE 0: O = acc + bias                        (first layer)
//    MODE 1: r = relu(acc+bias); O = r; fused per-column sum/sumsq reduction
//            (LDS ds_add_f32 per block, then one global atomic per column)
//    MODE 2: out[row, 2*col+1] = x[row, 2*col+1] + acc + bias   (final layer)
// ---------------------------------------------------------------------------
template <int MODE, int LDA, int KDIM, int NB>
__global__ __launch_bounds__(256) void gemm_wmma_kernel(
    const __bf16* __restrict__ A,
    const __bf16* __restrict__ Bp,
    const float*  __restrict__ bias,
    float* __restrict__ O,                          // MODE 0/1
    float* __restrict__ gstats,                     // MODE 1: [sum(1024), sumsq(1024)]
    const float* __restrict__ x,                    // MODE 2
    float* __restrict__ out)                        // MODE 2
{
    const int tid   = threadIdx.x;
    const int wave  = tid >> 5;          // 0..7
    const int lane  = tid & 31;
    const int waveM = wave >> 2;         // 0..1
    const int waveN = wave & 3;          // 0..3
    const int bm    = blockIdx.x * 128;
    const int bn    = blockIdx.y * 256;

    __shared__ float sSum[256];
    __shared__ float sSq[256];
    if (MODE == 1) {
        sSum[tid] = 0.0f;
        sSq[tid]  = 0.0f;
        __syncthreads();
    }

    v8f acc[4][4];
#pragma unroll
    for (int mt = 0; mt < 4; ++mt)
#pragma unroll
        for (int nt = 0; nt < 4; ++nt)
            acc[mt][nt] = (v8f)0.0f;

    const int laneRow   = lane & 15;
    const int laneKHalf = (lane >> 4) * 8;                    // 0 or 8
    const int rowBase   = bm + waveM * 64;
    const int colTile0  = (bn >> 4) + waveN * 4;              // global 16-col tile idx

    // Single A / B pointers; all tile offsets are compile-time immediates.
    const __bf16* pA = A + (size_t)(rowBase + laneRow) * LDA + laneKHalf;
    const __bf16* pB = Bp + ((size_t)colTile0 << 9) + lane * 16;

    for (int k = 0; k < KDIM; k += 32) {
        // --- A fragments (4 x 16x32 bf16), per-ISA lane layout ---
        v16bf afrag[4];
#pragma unroll
        for (int mt = 0; mt < 4; ++mt) {
            const __bf16* p = pA + mt * (16 * LDA);
            v8bf lo = *(const v8bf*)(p);        // K = k..k+7   (or +8..+15)
            v8bf hi = *(const v8bf*)(p + 16);   // K = k+16..23 (or +24..+31)
            v16bf a;
#pragma unroll
            for (int j = 0; j < 8; ++j) { a[j] = lo[j]; a[j + 8] = hi[j]; }
            afrag[mt] = a;
        }
        // --- B fragments: pre-packed, 32 contiguous bytes per lane ---
        v16bf bfrag[4];
#pragma unroll
        for (int nt = 0; nt < 4; ++nt)
            bfrag[nt] = *(const v16bf*)(pB + nt * 512);

#pragma unroll
        for (int nt = 0; nt < 4; ++nt)
#pragma unroll
            for (int mt = 0; mt < 4; ++mt)
                acc[mt][nt] = __builtin_amdgcn_wmma_f32_16x16x32_bf16(
                    false, afrag[mt], false, bfrag[nt],
                    (short)0, acc[mt][nt], false, false);

        pA += 32;
        pB += (size_t)NB * 512;
    }

    // --- epilogue ---
    const int cLane = lane & 15;
    const int rLane = (lane >> 4) * 8;   // C layout: lanes 16-31 hold rows +8
#pragma unroll
    for (int mt = 0; mt < 4; ++mt) {
#pragma unroll
        for (int nt = 0; nt < 4; ++nt) {
            int col  = bn + waveN * 64 + nt * 16 + cLane;
            int row0 = rowBase + mt * 16 + rLane;
            if (MODE == 0) {
                float bv = bias[col];
#pragma unroll
                for (int v = 0; v < 8; ++v)
                    O[(size_t)(row0 + v) * HID + col] = acc[mt][nt][v] + bv;
            } else if (MODE == 1) {
                float bv = bias[col];
                float s = 0.0f, q = 0.0f;
#pragma unroll
                for (int v = 0; v < 8; ++v) {
                    float r = acc[mt][nt][v] + bv;
                    r = fmaxf(r, 0.0f);
                    O[(size_t)(row0 + v) * HID + col] = r;
                    s += r;
                    q += r * r;
                }
                int cl = waveN * 64 + nt * 16 + cLane;
                atomicAdd(&sSum[cl], s);     // ds_add_f32
                atomicAdd(&sSq[cl], q);
            } else { // MODE 2
                if (col < HALF_D) {
                    float bv = bias[col];
#pragma unroll
                    for (int v = 0; v < 8; ++v) {
                        size_t idx = (size_t)(row0 + v) * DIN + 2 * col + 1;
                        out[idx] = x[idx] + acc[mt][nt][v] + bv;
                    }
                }
            }
        }
    }

    if (MODE == 1) {
        __syncthreads();
        atomicAdd(&gstats[bn + tid],       sSum[tid]);   // global_atomic_add_f32
        atomicAdd(&gstats[HID + bn + tid], sSq[tid]);
    }
}

// ---------------------------------------------------------------------------
// 4) Inter-layer pass: (optional batchnorm affine) + f32 -> bf16 for next GEMM.
// ---------------------------------------------------------------------------
__global__ __launch_bounds__(256) void convert_norm_kernel(
    const float* __restrict__ O, __bf16* __restrict__ A,
    const float* __restrict__ stats, const float* __restrict__ gamma,
    const float* __restrict__ beta, int useStats)
{
    size_t i = ((size_t)blockIdx.x * blockDim.x + threadIdx.x) * 4;
    float4 v = *(const float4*)(O + i);
    float r[4] = { v.x, v.y, v.z, v.w };
    if (useStats) {
        int col = (int)(i & (HID - 1));
#pragma unroll
        for (int j = 0; j < 4; ++j) {
            int c = col + j;
            float mu  = stats[c] * (1.0f / BATCH);
            float var = stats[HID + c] * (1.0f / BATCH) - mu * mu;
            float inv = rsqrtf(var + 1e-5f);
            r[j] = (r[j] - mu) * inv * gamma[c] + beta[c];
        }
    }
    v4bf o;
#pragma unroll
    for (int j = 0; j < 4; ++j) o[j] = (__bf16)r[j];
    *(v4bf*)(A + i) = o;
}

// ---------------------------------------------------------------------------
// Host launcher. Workspace layout (bytes):
//   Abuf  bf16 16384x1024           @ 0           (33,554,432)
//   Obuf  f32  16384x1024           @ 33,554,432  (67,108,864)
//   stats f32  2048                 @ 100,663,296 (8,192)
//   WinP  bf16 13x64 frags          @ 100,671,488 (851,968)
//   WhP   bf16 5 x 32x64 frags      @ 101,523,456 (10,485,760)
//   WoutP bf16 32x32 frags          @ 112,009,216 (1,048,576)   total ~113 MB
// ---------------------------------------------------------------------------
extern "C" void kernel_launch(void* const* d_in, const int* in_sizes, int n_in,
                              void* d_out, int out_size, void* d_ws, size_t ws_size,
                              hipStream_t stream)
{
    const float* x     = (const float*)d_in[0];
    const float* Win   = (const float*)d_in[1];
    const float* bin_  = (const float*)d_in[2];
    const float* Wh    = (const float*)d_in[3];
    const float* bh    = (const float*)d_in[4];
    const float* gamma = (const float*)d_in[5];
    const float* beta  = (const float*)d_in[6];
    const float* Wout  = (const float*)d_in[7];
    const float* bout  = (const float*)d_in[8];
    float* out = (float*)d_out;

    char* ws = (char*)d_ws;
    __bf16* Abuf  = (__bf16*)(ws);
    float*  Obuf  = (float*)(ws + 33554432);
    float*  stats = (float*)(ws + 100663296);
    __bf16* WinP  = (__bf16*)(ws + 100671488);
    __bf16* WhP   = (__bf16*)(ws + 101523456);
    __bf16* WoutP = (__bf16*)(ws + 112009216);

    // x1 passthrough + bf16 A matrix (padded K)
    pack_x_kernel<<<(BATCH * K0PAD) / 256, 256, 0, stream>>>(x, out, Abuf);

    // Weight packing (bf16 WMMA fragments)
    pack_w_kernel<<<(13 * 64 * 32) / 256, 256, 0, stream>>>(
        Win, WinP, HALF_D, HID, 13, 64);
    for (int i = 0; i < 5; ++i) {
        pack_w_kernel<<<(32 * 64 * 32) / 256, 256, 0, stream>>>(
            Wh + (size_t)i * HID * HID, WhP + (size_t)i * 1048576, HID, HID, 32, 64);
    }
    pack_w_kernel<<<(32 * 32 * 32) / 256, 256, 0, stream>>>(
        Wout, WoutP, HID, HALF_D, 32, 32);

    dim3 grid(BATCH / 128, HID / 256);   // 128 x 4

    // Layer 0: h = x1 @ Win + bin_
    gemm_wmma_kernel<0, K0PAD, K0PAD, 64><<<grid, 256, 0, stream>>>(
        Abuf, WinP, bin_, Obuf, nullptr, nullptr, nullptr);
    convert_norm_kernel<<<(BATCH * HID) / (256 * 4), 256, 0, stream>>>(
        Obuf, Abuf, nullptr, nullptr, nullptr, 0);

    // 5 hidden layers: relu(h @ Wh + bh) -> batchnorm
    for (int i = 0; i < 5; ++i) {
        zero_stats_kernel<<<8, 256, 0, stream>>>(stats);
        gemm_wmma_kernel<1, HID, HID, 64><<<grid, 256, 0, stream>>>(
            Abuf, WhP + (size_t)i * 1048576, bh + (size_t)i * HID,
            Obuf, stats, nullptr, nullptr);
        convert_norm_kernel<<<(BATCH * HID) / (256 * 4), 256, 0, stream>>>(
            Obuf, Abuf, stats, gamma + (size_t)i * HID, beta + (size_t)i * HID, 1);
    }

    // Output layer: y1 = x_odd + h @ Wout + bout, interleaved store
    dim3 grid2(BATCH / 128, NOUTPAD / 256);   // 128 x 2
    gemm_wmma_kernel<2, HID, HID, 32><<<grid2, 256, 0, stream>>>(
        Abuf, WoutP, bout, nullptr, nullptr, x, out);
}
